// ConvNet_2448131358723
// MI455X (gfx1250) — compile-verified, matrix operand order checked by reading, not run on hardware
//
#include <hip/hip_runtime.h>
#include <hip/hip_bf16.h>

// ---------------------------------------------------------------------------
// ConvNet text classifier on CDNA5 (gfx1250), bf16 WMMA with f32 accumulate.
// W=4096 words, L=32 chars, D=256.
// ---------------------------------------------------------------------------

typedef __attribute__((ext_vector_type(16))) __bf16 v16bf;
typedef __attribute__((ext_vector_type(8)))  float  v8f;

__device__ __forceinline__ unsigned short f2bf(float f) {
  // round-to-nearest-even fp32 -> bf16
  unsigned int x = __float_as_uint(f);
  unsigned int r = x + 0x7fffu + ((x >> 16) & 1u);
  return (unsigned short)(r >> 16);
}

union FragU { uint4 q[2]; v16bf v; };

// Load one 16x32 bf16 A/B fragment chunk-pair. Per ISA 7.12.2 (16-bit A):
// lanes 0-15 hold K = [h0 .. h0+7] and [h0+16 .. h0+23] of their row,
// lanes 16-31 hold the +8 shifted halves (caller folds that into h0 via koff).
__device__ __forceinline__ v16bf load_frag(const unsigned short* rowp, int h0) {
  FragU u;
  u.q[0] = *(const uint4*)(rowp + h0);
  u.q[1] = *(const uint4*)(rowp + h0 + 16);
  return u.v;
}

__device__ __forceinline__ v16bf zero_frag() {
  FragU u;
  u.q[0] = make_uint4(0u, 0u, 0u, 0u);
  u.q[1] = make_uint4(0u, 0u, 0u, 0u);
  return u.v;
}

#define WMMA_BF16(A, B, C) \
  __builtin_amdgcn_wmma_f32_16x16x32_bf16(false, (A), false, (B), (short)0, (C), false, false)

// ---------------------------------------------------------------------------
// Kernel 1: weight transposition to bf16 im2col-friendly layout + word-emb
// gather into the first half of u.
//   BcT[o][k*256+i] = conv_chr_w[o][i][k]      (256 x 768)
//   BsT[o][k*512+i] = conv_sent_w[o][i][k]     (512 x 1536)
//   u[w][0:256]     = bf16(word_emb[words[w]]) (4096 x 512, char half later)
// ---------------------------------------------------------------------------
__global__ __launch_bounds__(256) void prep_kernel(
    const float* __restrict__ cw, const float* __restrict__ sw,
    const int* __restrict__ words, const float* __restrict__ word_emb,
    unsigned short* __restrict__ BcT, unsigned short* __restrict__ BsT,
    unsigned short* __restrict__ u) {
  int gid = blockIdx.x * 256 + threadIdx.x;
  const int NC = 256 * 768;    // 196608
  const int NS = 512 * 1536;   // 786432
  const int NU = 4096 * 256;   // 1048576
  if (gid < NC) {
    int o = gid / 768;
    int kk = gid - o * 768;
    int k = kk >> 8, i = kk & 255;
    BcT[gid] = f2bf(cw[(o * 256 + i) * 3 + k]);
  } else if (gid < NC + NS) {
    int g = gid - NC;
    int o = g / 1536;
    int kk = g - o * 1536;
    int k = kk / 512, i = kk - k * 512;
    BsT[g] = f2bf(sw[(o * 512 + i) * 3 + k]);
  } else if (gid < NC + NS + NU) {
    int g = gid - (NC + NS);
    int w = g >> 8, c = g & 255;
    u[w * 512 + c] = f2bf(word_emb[words[w] * 256 + c]);
  }
}

// ---------------------------------------------------------------------------
// Kernel 2: char conv + maxpool, one block (8 waves) per word.
//   im2col A[t][k*256+i] = chr_emb[wic[w][t+k-1]][i] staged in LDS (bf16).
//   Y[32,256] = A[32,768] @ BcT^T via 2 M-tiles x 2 N-tiles/wave x 24 K-steps.
//   u[w][256+o] = bf16(max_t Y[t][o] + bias[o]).
// ---------------------------------------------------------------------------
__global__ __launch_bounds__(256) void char_conv_kernel(
    const int* __restrict__ wic, const float* __restrict__ chr_emb,
    const unsigned short* __restrict__ BcT, const float* __restrict__ bias,
    unsigned short* __restrict__ u) {
  __shared__ unsigned short A[32 * 768];  // 48 KB

  const int word = blockIdx.x;
  const int tid  = threadIdx.x;
  const int* idxrow = wic + word * 32;

  // im2col staging (zero padded at t-1 < 0, t+1 > 31)
  for (int e = tid; e < 32 * 768; e += 256) {
    int t = e / 768;
    int kk = e - t * 768;
    int k = kk >> 8, i = kk & 255;
    int st = t + k - 1;
    unsigned short v = 0;
    if (st >= 0 && st < 32) v = f2bf(chr_emb[idxrow[st] * 256 + i]);
    A[e] = v;
  }
  __syncthreads();

  const int lane = tid & 31;
  const int wv   = tid >> 5;           // 8 waves
  const int m    = lane & 15;
  const int koff = (lane >> 4) << 3;   // 0 for lanes 0-15, 8 for lanes 16-31

  const unsigned short* a0p = A + m * 768;          // M-tile 0 rows 0..15
  const unsigned short* a1p = A + (m + 16) * 768;   // M-tile 1 rows 16..31
  const int n0 = wv * 32 + m;                       // N-tile 2*wv
  const int n1 = n0 + 16;                           // N-tile 2*wv+1
  const unsigned short* b0p = BcT + n0 * 768;
  const unsigned short* b1p = BcT + n1 * 768;

  v8f c00 = {0.f,0.f,0.f,0.f,0.f,0.f,0.f,0.f};
  v8f c01 = c00, c10 = c00, c11 = c00;

  for (int kb = 0; kb < 24; ++kb) {
    int h0 = kb * 32 + koff;
    v16bf a0 = load_frag(a0p, h0);
    v16bf a1 = load_frag(a1p, h0);
    v16bf b0 = load_frag(b0p, h0);
    v16bf b1 = load_frag(b1p, h0);
    c00 = WMMA_BF16(a0, b0, c00);
    c01 = WMMA_BF16(a0, b1, c01);
    c10 = WMMA_BF16(a1, b0, c10);
    c11 = WMMA_BF16(a1, b1, c11);
  }

  // maxpool over t: VGPR r holds row (lane<16 ? r : 8+r) per M-tile, col = lane&15.
  float m0 = -3.0e38f, m1 = -3.0e38f;
  #pragma unroll
  for (int r = 0; r < 8; ++r) {
    m0 = fmaxf(m0, fmaxf(c00[r], c10[r]));
    m1 = fmaxf(m1, fmaxf(c01[r], c11[r]));
  }
  m0 = fmaxf(m0, __shfl_xor(m0, 16));
  m1 = fmaxf(m1, __shfl_xor(m1, 16));
  if (lane < 16) {
    u[word * 512 + 256 + n0] = f2bf(m0 + bias[n0]);
    u[word * 512 + 256 + n1] = f2bf(m1 + bias[n1]);
  }
}

// ---------------------------------------------------------------------------
// Kernel 3: sentence conv + partial maxpool over the word axis.
// 32 blocks x 8 waves = 256 waves; wave gw owns rows [gw*16, gw*16+16).
// Three shifted GEMMs (k=0..2), K=512 each, 32 N-tiles, f32 accumulate.
// partials[gw][o] = max over this wave's 16 rows.
// ---------------------------------------------------------------------------
__global__ __launch_bounds__(256) void sent_conv_kernel(
    const unsigned short* __restrict__ u, const unsigned short* __restrict__ BsT,
    float* __restrict__ partials) {
  const int tid  = threadIdx.x;
  const int lane = tid & 31;
  const int wv   = tid >> 5;
  const int gw   = blockIdx.x * 8 + wv;  // m-tile id, 0..255
  const int m    = lane & 15;
  const int koff = (lane >> 4) << 3;

  for (int nt = 0; nt < 32; ++nt) {
    const int n = nt * 16 + m;
    const unsigned short* bp = BsT + n * 1536;
    v8f c = {0.f,0.f,0.f,0.f,0.f,0.f,0.f,0.f};
    #pragma unroll
    for (int k = 0; k < 3; ++k) {
      int trow = gw * 16 + m + k - 1;
      bool valid = (trow >= 0) && (trow < 4096);
      const unsigned short* ap = u + (valid ? trow : 0) * 512;
      for (int kb = 0; kb < 16; ++kb) {
        int h0 = kb * 32 + koff;
        v16bf a = zero_frag();
        if (valid) a = load_frag(ap, h0);
        v16bf b = load_frag(bp, k * 512 + h0);
        c = WMMA_BF16(a, b, c);
      }
    }
    float v = c[0];
    #pragma unroll
    for (int r = 1; r < 8; ++r) v = fmaxf(v, c[r]);
    v = fmaxf(v, __shfl_xor(v, 16));
    if (lane < 16) partials[gw * 512 + n] = v;
  }
}

// ---------------------------------------------------------------------------
// Kernel 4: final max-reduce + bias + tanh FC head -> 2 logits.
// ---------------------------------------------------------------------------
__global__ __launch_bounds__(1024) void head_kernel(
    const float* __restrict__ partials, const float* __restrict__ sent_b,
    const float* __restrict__ w1, const float* __restrict__ b1,
    const float* __restrict__ w2, const float* __restrict__ b2,
    float* __restrict__ out) {
  __shared__ float r_s[512];
  __shared__ float h_s[1024];
  const int tid = threadIdx.x;
  if (tid < 512) {
    float mx = -3.0e38f;
    for (int g = 0; g < 256; ++g) mx = fmaxf(mx, partials[g * 512 + tid]);
    r_s[tid] = mx + sent_b[tid];
  }
  __syncthreads();
  {
    float acc = b1[tid];
    const float* wr = w1 + tid * 512;
    for (int c = 0; c < 512; ++c) acc += wr[c] * r_s[c];
    h_s[tid] = tanhf(acc);
  }
  __syncthreads();
  if (tid < 2) {
    float acc = b2[tid];
    const float* wr = w2 + tid * 1024;
    for (int j = 0; j < 1024; ++j) acc += wr[j] * h_s[j];
    out[tid] = acc;
  }
}

// ---------------------------------------------------------------------------
// Host launcher.
// ---------------------------------------------------------------------------
extern "C" void kernel_launch(void* const* d_in, const int* in_sizes, int n_in,
                              void* d_out, int out_size, void* d_ws, size_t ws_size,
                              hipStream_t stream) {
  (void)in_sizes; (void)n_in; (void)out_size; (void)ws_size;

  const int*   words       = (const int*)  d_in[0];
  const int*   wic         = (const int*)  d_in[1];
  const float* word_emb    = (const float*)d_in[2];
  const float* chr_emb     = (const float*)d_in[3];
  const float* conv_chr_w  = (const float*)d_in[4];
  const float* conv_chr_b  = (const float*)d_in[5];
  const float* conv_sent_w = (const float*)d_in[6];
  const float* conv_sent_b = (const float*)d_in[7];
  const float* w1          = (const float*)d_in[8];
  const float* b1          = (const float*)d_in[9];
  const float* w2          = (const float*)d_in[10];
  const float* b2          = (const float*)d_in[11];
  float* out = (float*)d_out;

  char* ws = (char*)d_ws;
  unsigned short* u        = (unsigned short*)(ws + 0);        // 4096*512*2   = 4,194,304
  unsigned short* BcT      = (unsigned short*)(ws + 4194304);  // 256*768*2    =   393,216
  unsigned short* BsT      = (unsigned short*)(ws + 4587520);  // 512*1536*2   = 1,572,864
  float*          partials = (float*)        (ws + 6160384);   // 256*512*4    =   524,288

  // 196608 + 786432 + 1048576 = 2031616 elements = 7936 blocks of 256
  prep_kernel<<<7936, 256, 0, stream>>>(conv_chr_w, conv_sent_w, words, word_emb,
                                        BcT, BsT, u);
  char_conv_kernel<<<4096, 256, 0, stream>>>(wic, chr_emb, BcT, conv_chr_b, u);
  sent_conv_kernel<<<32, 256, 0, stream>>>(u, BsT, partials);
  head_kernel<<<1, 1024, 0, stream>>>(partials, conv_sent_b, w1, b1, w2, b2, out);
}
